// AFM_87342454931962
// MI455X (gfx1250) — matrix operand order repaired and checked by reading, not compiled, not run on hardware
//
#include <hip/hip_runtime.h>
#include <math.h>

// ---------------------------------------------------------------------------
// AFM forward for MI455X (gfx1250, wave32).
//
// Dominant cost: per-sample GEMM emb_cross[741,32] @ att_w_W[32,64]
//   -> 25 GFLOP total, mapped onto V_WMMA_F32_16X16X32_F16 (K=EMB=32 exact).
// Memory: ~41 MB of gathered embeddings -> ~2us at 23.3 TB/s; compute-bound
// unless matrix pipes are used, hence f16 WMMA with f32 accumulation.
// One workgroup (256 thr = 8 waves) per batch element; all tensor staging in
// LDS (~20 KB of the 320 KB/WGP), so many blocks co-reside per WGP.
//
// Two pair-tiles are processed per wave iteration so the second tile's
// A-fragment build + WMMAs fill the first tile's WMMA->VALU hazard window
// (4 co-exec slots for f16 WMMA) instead of compiler-inserted v_nops.
// ---------------------------------------------------------------------------

typedef __attribute__((ext_vector_type(16))) _Float16 v16h;
typedef __attribute__((ext_vector_type(8)))  float    v8f;

#define FIELD   39
#define EMB     32
#define ATT     64
#define NPAIR   741            // 39*38/2
#define NTILE   47             // ceil(741/16); 47*16 = 752
#define NPAD    (NTILE * 16)
#define BLOCK   256
#define NWAVE   8

__global__ __launch_bounds__(BLOCK)
void afm_fwd_kernel(const int* __restrict__ idxs,       // [B,39]
                    const float* __restrict__ vals,     // [B,39]
                    const float* __restrict__ fm_bias,  // [1]
                    const float* __restrict__ first_w,  // [1M,1]
                    const float* __restrict__ fm_emb,   // [1M,32]
                    const float* __restrict__ att_w_W,  // [32,64]
                    const float* __restrict__ att_w_b,  // [64]
                    const float* __restrict__ att_b,    // [64]
                    const float* __restrict__ att_h_W,  // [64,1]
                    const float* __restrict__ att_h_b,  // [1]
                    const float* __restrict__ fc_p_W,   // [32,1]
                    const float* __restrict__ fc_p_b,   // [1]
                    float* __restrict__ out)            // [B]
{
    __shared__ float          s_emb[FIELD][EMB];  // scaled embeddings
    __shared__ float          s_W[EMB * ATT];     // att_w_W staged
    __shared__ float          s_wb[ATT];
    __shared__ float          s_ab[ATT];
    __shared__ float          s_hw[ATT];
    __shared__ unsigned short s_pair[NPAD];       // (row<<8)|col
    __shared__ float          s_att[NPAD];        // att_e -> exp(att_e - max)
    __shared__ float          s_red[BLOCK];
    __shared__ float          s_sum_emb[EMB];
    __shared__ float          s_first;

    const int b      = blockIdx.x;
    const int tid    = threadIdx.x;
    const int lane   = tid & 31;
    const int wave   = tid >> 5;
    const int lane16 = lane & 15;
    const int khalf  = lane >> 4;          // 0: lanes 0-15, 1: lanes 16-31

    // ---------------- Phase 0: stage weights, pair table, accumulators -----
    if (tid < EMB) s_sum_emb[tid] = 0.0f;
    if (tid == 0)  s_first = 0.0f;

    for (int i = tid; i < EMB * ATT; i += BLOCK) s_W[i] = att_w_W[i];
    if (tid < ATT) {
        s_wb[tid] = att_w_b[tid];
        s_ab[tid] = att_b[tid];
        s_hw[tid] = att_h_W[tid];
    }

    for (int p = tid; p < NPAD; p += BLOCK) {
        unsigned short rc = 0;
        if (p < NPAIR) {
            int pp = p, r = 0;
            while (pp >= (FIELD - 1 - r)) { pp -= (FIELD - 1 - r); ++r; }
            rc = (unsigned short)((r << 8) | (r + 1 + pp));
        }
        s_pair[p] = rc;
    }

    // ---------------- Phase 1: gather + scale embeddings (B128 loads) ------
    const int*   bidx = idxs + (long)b * FIELD;
    const float* bval = vals + (long)b * FIELD;
    for (int i = tid; i < FIELD * 8; i += BLOCK) {        // 8 float4 per row
        const int f  = i >> 3;
        const int ch = i & 7;
        const float v = bval[f];
        const float4 e = ((const float4*)(fm_emb + (long)bidx[f] * EMB))[ch];
        float4 r;
        r.x = e.x * v; r.y = e.y * v; r.z = e.z * v; r.w = e.w * v;
        ((float4*)&s_emb[f][0])[ch] = r;
    }

    // first-order term (39 scalar gathers, ds_add_f32 reduction)
    if (tid < FIELD) {
        atomicAdd(&s_first, first_w[bidx[tid]] * bval[tid]);
    }

    __syncthreads();

    // ---------------- Phase 2: attention logits via WMMA --------------------
    // B fragment (att_w_W tile, 32x16 f16): half i of lane -> K = khalf*16+i,
    // N = t4*16 + lane16  (contiguous k, same order as A).
    v16h bfrag[4];
#pragma unroll
    for (int t4 = 0; t4 < 4; ++t4) {
#pragma unroll
        for (int i = 0; i < 16; ++i) {
            const int k = khalf * 16 + i;
            bfrag[t4][i] = (_Float16)s_W[k * ATT + t4 * 16 + lane16];
        }
    }

    const float att_hb = att_h_b[0];

    // A fragment (emb_cross tile, 16x32 f16). Row m = lane16,
    // half i -> K = khalf*16 + i  (contiguous, matches B fragment).
    auto build_a = [&](int t) -> v16h {
        const int p0 = t * 16 + lane16;
        const int rc = s_pair[p0];
        const int rI = rc >> 8, cI = rc & 255;
        v16h a;
#pragma unroll
        for (int i = 0; i < 16; ++i) {
            const int k = khalf * 16 + i;
            a[i] = (_Float16)(s_emb[rI][k] * s_emb[cI][k]);
        }
        return a;
    };

    // Fused epilogue: ReLU + biases + att_h dot, then reduce across the 16
    // N-lanes of each half-wave (xor 1,2,4,8 stays inside 16-lane groups).
    // C layout: VGPR j -> row (j + 8*khalf), col n = t4*16 + lane16.
    auto epilogue = [&](const v8f* cacc, int t) {
        float ep[8];
#pragma unroll
        for (int j = 0; j < 8; ++j) ep[j] = 0.0f;
#pragma unroll
        for (int t4 = 0; t4 < 4; ++t4) {
            const int   n   = t4 * 16 + lane16;
            const float wbn = s_wb[n];
            const float abn = s_ab[n];
            const float hwn = s_hw[n];
#pragma unroll
            for (int j = 0; j < 8; ++j) {
                ep[j] += (fmaxf(cacc[t4][j] + wbn, 0.0f) + abn) * hwn;
            }
        }
#pragma unroll
        for (int s = 1; s < 16; s <<= 1) {
#pragma unroll
            for (int j = 0; j < 8; ++j) ep[j] += __shfl_xor(ep[j], s);
        }
        // lanes 0-7 own rows 0-7, lanes 16-23 own rows 8-15
        if (lane16 < 8) {
            s_att[t * 16 + lane16 + 8 * khalf] = ep[lane16] + att_hb;
        }
    };

    // Wave-uniform dual-tile loop: pairs (t, t+8) at stride 16 cover all 47
    // tiles; the trailing guard is wave-uniform so EXEC is all-1s at every
    // WMMA. The second tile's A-build/WMMAs overlap the first tile's
    // WMMA->VALU hazard window.
    for (int t = wave; t < NTILE; t += 2 * NWAVE) {
        const int t2 = t + NWAVE;

        const v16h a0 = build_a(t);
        v8f c0[4];
#pragma unroll
        for (int t4 = 0; t4 < 4; ++t4) {
            v8f z = {};
            c0[t4] = __builtin_amdgcn_wmma_f32_16x16x32_f16(
                false, a0, false, bfrag[t4], (short)0, z, false, false);
        }

        if (t2 < NTILE) {
            const v16h a1 = build_a(t2);
            v8f c1[4];
#pragma unroll
            for (int t4 = 0; t4 < 4; ++t4) {
                v8f z = {};
                c1[t4] = __builtin_amdgcn_wmma_f32_16x16x32_f16(
                    false, a1, false, bfrag[t4], (short)0, z, false, false);
            }
            epilogue(c0, t);
            epilogue(c1, t2);
        } else {
            epilogue(c0, t);
        }
    }
    __syncthreads();

    // ---------------- Phase 3: softmax over 741 pairs -----------------------
    float lmax = -3.4e38f;
    for (int p = tid; p < NPAIR; p += BLOCK) lmax = fmaxf(lmax, s_att[p]);
    s_red[tid] = lmax;
    __syncthreads();
#pragma unroll
    for (int s = BLOCK / 2; s > 0; s >>= 1) {
        if (tid < s) s_red[tid] = fmaxf(s_red[tid], s_red[tid + s]);
        __syncthreads();
    }
    const float gmax = s_red[0];
    __syncthreads();

    float lsum = 0.0f;
    for (int p = tid; p < NPAIR; p += BLOCK) {
        const float e = __expf(s_att[p] - gmax);
        s_att[p] = e;
        lsum += e;
    }
    s_red[tid] = lsum;
    __syncthreads();
#pragma unroll
    for (int s = BLOCK / 2; s > 0; s >>= 1) {
        if (tid < s) s_red[tid] += s_red[tid + s];
        __syncthreads();
    }
    const float inv_sum = 1.0f / s_red[0];

    // ---------------- Phase 4: attention-weighted sum over pairs ------------
    {
        const int e   = tid & (EMB - 1);
        const int grp = tid >> 5;                       // 8 pair-groups
        float acc = 0.0f;
        for (int p = grp; p < NPAIR; p += NWAVE) {
            const int rc = s_pair[p];
            const int rI = rc >> 8, cI = rc & 255;
            acc += s_att[p] * s_emb[rI][e] * s_emb[cI][e];
        }
        atomicAdd(&s_sum_emb[e], acc);                  // ds_add_f32
    }
    __syncthreads();

    // ---------------- Phase 5: projection + sigmoid (wave 0) ----------------
    if (wave == 0) {
        float v = (lane < EMB) ? s_sum_emb[lane] * fc_p_W[lane] : 0.0f;
#pragma unroll
        for (int s = 16; s > 0; s >>= 1) v += __shfl_xor(v, s);
        if (lane == 0) {
            const float second = v * inv_sum + fc_p_b[0];
            const float x = fm_bias[0] + s_first + second;
            out[b] = 1.0f / (1.0f + __expf(-x));
        }
    }
}

extern "C" void kernel_launch(void* const* d_in, const int* in_sizes, int n_in,
                              void* d_out, int out_size, void* d_ws, size_t ws_size,
                              hipStream_t stream) {
    (void)n_in; (void)d_ws; (void)ws_size;
    const int*   idxs    = (const int*)  d_in[0];
    const float* vals    = (const float*)d_in[1];
    const float* fm_bias = (const float*)d_in[2];
    const float* first_w = (const float*)d_in[3];
    const float* fm_emb  = (const float*)d_in[4];
    const float* att_w_W = (const float*)d_in[5];
    const float* att_w_b = (const float*)d_in[6];
    const float* att_b   = (const float*)d_in[7];
    const float* att_h_W = (const float*)d_in[8];
    const float* att_h_b = (const float*)d_in[9];
    const float* fc_p_W  = (const float*)d_in[10];
    const float* fc_p_b  = (const float*)d_in[11];
    float* out = (float*)d_out;

    const int batch = out_size;              // 8192
    hipLaunchKernelGGL(afm_fwd_kernel, dim3(batch), dim3(BLOCK), 0, stream,
                       idxs, vals, fm_bias, first_w, fm_emb, att_w_W,
                       att_w_b, att_b, att_h_W, att_h_b, fc_p_W, fc_p_b, out);
}
